// OuterProductMean_59476707115314
// MI455X (gfx1250) — matrix operand order repaired and checked by reading, not compiled, not run on hardware
//
#include <hip/hip_runtime.h>
#include <hip/hip_bf16.h>

// ---------------------------------------------------------------------------
// OuterProductMean for MI455X (gfx1250, wave32, WMMA 16x16x32 f16,
// async global->LDS double-buffered staging)
// ---------------------------------------------------------------------------
typedef __attribute__((ext_vector_type(16))) _Float16 v16h;
typedef __attribute__((ext_vector_type(8)))  _Float16 v8h;
typedef __attribute__((ext_vector_type(8)))  float    v8f;
typedef __attribute__((ext_vector_type(4)))  float    v4f;

static constexpr int NSEQ = 512;   // a
static constexpr int NRES = 384;   // b, d
static constexpr int CM   = 256;   // m
static constexpr int CZ   = 128;   // f

// workspace layout (bytes)
static constexpr size_t OFF_L    = 0;                          // [12288][512] f16
static constexpr size_t OFF_R    = OFF_L + (size_t)12288*512*2;// [12288][512] f16
static constexpr size_t OFF_WLR  = OFF_R + (size_t)12288*512*2;// [64][256]   f16
static constexpr size_t OFF_WO   = OFF_WLR + (size_t)64*256*2; // [128][1024] f16
static constexpr size_t OFF_NORM = OFF_WO + (size_t)128*1024*2;// [384][384]  f32

// ---------------------------------------------------------------------------
// Prep: transpose weights into [N][K] f16 layouts (B-matrix friendly)
// ---------------------------------------------------------------------------
__global__ void k_prep(const float* __restrict__ lw, const float* __restrict__ rw,
                       const float* __restrict__ ow,
                       _Float16* __restrict__ Wlr, _Float16* __restrict__ Wo) {
  int i = blockIdx.x * blockDim.x + threadIdx.x;
  if (i < 64 * 256) {
    int n = i >> 8, k = i & 255;
    float v = (n < 32) ? lw[k * 32 + n] : rw[k * 32 + (n - 32)];
    Wlr[n * 256 + k] = (_Float16)v;
  }
  int j = i - 64 * 256;
  if (j >= 0 && j < 128 * 1024) {
    int f = j >> 10, ce = j & 1023;           // ce = c*32 + e
    Wo[f * 1024 + ce] = (_Float16)ow[(size_t)ce * 128 + f];
  }
}

// ---------------------------------------------------------------------------
// Mask Gram matrix: norm[b][d] = sum_a mask[a,b]*mask[a,d]
// ---------------------------------------------------------------------------
__global__ void k_norm(const float* __restrict__ mask, float* __restrict__ nrm) {
  int p = blockIdx.x * blockDim.x + threadIdx.x;
  if (p >= NRES * NRES) return;
  int b = p / NRES, d = p % NRES;
  float s = 0.f;
  for (int a = 0; a < NSEQ; ++a)
    s += mask[a * NRES + b] * mask[a * NRES + d];
  nrm[p] = s;
}

// ---------------------------------------------------------------------------
// LayerNorm fused with projection GEMM (WMMA).
// grid (8 a-chunks, 384 b), 128 threads = 4 waves. Per WG: M=64 rows of a,
// N=64 (32 left | 32 right), K=256.  Output: masked f16 into
// Lbuf[(b*32+c)][a], Rbuf[(b*32+e)][a]  (GEMM1 A/B layouts).
// ---------------------------------------------------------------------------
__global__ __launch_bounds__(128) void k_ln_proj(
    const float* __restrict__ act, const float* __restrict__ mask,
    const float* __restrict__ ln_scale, const float* __restrict__ ln_offset,
    const float* __restrict__ left_b, const float* __restrict__ right_b,
    const _Float16* __restrict__ Wlr,
    _Float16* __restrict__ Lbuf, _Float16* __restrict__ Rbuf)
{
  __shared__ _Float16 As[64 * 264];     // row stride 264 halfs (528B, 16B-mult, conflict-free)
  __shared__ float sScale[CM], sOff[CM];

  const int tid  = threadIdx.x;
  const int lane = tid & 31, wave = tid >> 5;
  const int lsub = lane & 15, hi = lane >> 4;
  const int a0   = blockIdx.x * 64;
  const int b    = blockIdx.y;

  for (int i = tid; i < CM; i += 128) { sScale[i] = ln_scale[i]; sOff[i] = ln_offset[i]; }
  __syncthreads();

  // ---- phase 1: LayerNorm 16 rows per wave, wave-shuffle reduction ----
  for (int r = 0; r < 16; ++r) {
    int row = wave * 16 + r;
    int a   = a0 + row;
    const float* src = act + ((size_t)a * NRES + b) * CM + lane * 8;
    v4f x0 = *(const v4f*)(src);
    v4f x1 = *(const v4f*)(src + 4);
    float s = 0.f, ss = 0.f;
#pragma unroll
    for (int j = 0; j < 4; ++j) {
      s  += x0[j] + x1[j];
      ss += x0[j] * x0[j] + x1[j] * x1[j];
    }
#pragma unroll
    for (int off = 16; off >= 1; off >>= 1) {
      s  += __shfl_xor(s,  off, 32);
      ss += __shfl_xor(ss, off, 32);
    }
    float mu   = s * (1.0f / CM);
    float var  = ss * (1.0f / CM) - mu * mu;
    float rstd = rsqrtf(var + 1e-5f);
    int ch = lane * 8;
    v8h h;
#pragma unroll
    for (int j = 0; j < 8; ++j) {
      float xv = (j < 4) ? x0[j] : x1[j - 4];
      h[j] = (_Float16)((xv - mu) * rstd * sScale[ch + j] + sOff[ch + j]);
    }
    *(v8h*)(&As[row * 264 + ch]) = h;
  }
  __syncthreads();

  // ---- phase 2: WMMA projection. wave w -> M-tile w; 4 N-tiles; K=256 ----
  v8f acc[4];
#pragma unroll
  for (int nt = 0; nt < 4; ++nt) acc[nt] = (v8f){};
  const _Float16* arow = &As[(wave * 16 + lsub) * 264];

  for (int kc = 0; kc < 8; ++kc) {
    v16h afrag;
    const _Float16* ap = arow + kc * 32 + hi * 8;
    ((v8h*)&afrag)[0] = *(const v8h*)(ap);
    ((v8h*)&afrag)[1] = *(const v8h*)(ap + 16);
#pragma unroll
    for (int nt = 0; nt < 4; ++nt) {
      const _Float16* bp = Wlr + (nt * 16 + lsub) * 256 + kc * 32 + hi * 8;
      v16h bfrag;
      ((v8h*)&bfrag)[0] = *(const v8h*)(bp);
      ((v8h*)&bfrag)[1] = *(const v8h*)(bp + 16);
      acc[nt] = __builtin_amdgcn_wmma_f32_16x16x32_f16(
          false, afrag, false, bfrag, (short)0, acc[nt], false, false);
    }
  }

  // ---- epilogue: bias + mask, scatter f16 into GEMM1 layouts ----
#pragma unroll
  for (int nt = 0; nt < 4; ++nt) {
    int n = nt * 16 + lsub;
    float bias = (n < 32) ? left_b[n] : right_b[n - 32];
    _Float16* dst = (n < 32) ? (Lbuf + (size_t)(b * 32 + n) * NSEQ)
                             : (Rbuf + (size_t)(b * 32 + (n - 32)) * NSEQ);
#pragma unroll
    for (int r = 0; r < 8; ++r) {
      int a = a0 + wave * 16 + r + hi * 8;          // C-layout: VGPR r -> M=r / r+8
      float mv = mask[(size_t)a * NRES + b];
      dst[a] = (_Float16)((acc[nt][r] + bias) * mv);
    }
  }
}

// ---------------------------------------------------------------------------
// Fused outer-product GEMM1 + projection GEMM2.
// grid (48,48), 1024 threads = 32 waves. Per WG: (b,d) 8x8 pair block.
// GEMM1: [256 x 256 x 512] f16 WMMA, staging via double-buffered
//        GLOBAL_LOAD_ASYNC_TO_LDS_B128 (ASYNCcnt) -> acc tiles -> LDS inter.
// GEMM2: [64 pairs x 128 f x 1024 ce] WMMA, epilogue bias + /(eps+norm).
// ---------------------------------------------------------------------------
__global__ __launch_bounds__(1024) void k_opm(
    const _Float16* __restrict__ Lbuf, const _Float16* __restrict__ Rbuf,
    const _Float16* __restrict__ Wo, const float* __restrict__ nrm,
    const float* __restrict__ out_b, float* __restrict__ out)
{
  extern __shared__ __align__(16) char smem[];
  // double-buffered staging: A0|A1|B0|B1, each 256 rows x 40 halfs (80B rows)
  _Float16* Abuf0 = (_Float16*)smem;
  _Float16* Abuf1 = Abuf0 + 256 * 40;
  _Float16* Bbuf0 = Abuf1 + 256 * 40;
  _Float16* Bbuf1 = Bbuf0 + 256 * 40;
  _Float16* inter = (_Float16*)smem;          // aliased after GEMM1: 256 x 264 halfs

  const int tid  = threadIdx.x;
  const int lane = tid & 31, wave = tid >> 5;
  const int lsub = lane & 15, hi = lane >> 4;
  const int bb = blockIdx.x, dd = blockIdx.y;
  const size_t m0 = (size_t)bb * 256, n0 = (size_t)dd * 256;

  // GEMM1 wave tiling: 4 (M groups of 4 tiles) x 8 (N groups of 2 tiles)
  const int mg = wave >> 3, ng = wave & 7;
  v8f acc[4][2];
#pragma unroll
  for (int im = 0; im < 4; ++im)
#pragma unroll
    for (int j = 0; j < 2; ++j) acc[im][j] = (v8f){};

  // staging mapping: 1024 threads = 256 rows x 4 x 16B per buffer
  const int srow = tid >> 2;
  const int spos = tid & 3;
  const _Float16* gA = Lbuf + (m0 + srow) * NSEQ + spos * 8;
  const _Float16* gB = Rbuf + (n0 + srow) * NSEQ + spos * 8;
  const unsigned stageOff = (unsigned)((srow * 40 + spos * 8) * 2);   // bytes
  const unsigned ldsA[2] = { (unsigned)(uintptr_t)Abuf0 + stageOff,
                             (unsigned)(uintptr_t)Abuf1 + stageOff };
  const unsigned ldsB[2] = { (unsigned)(uintptr_t)Bbuf0 + stageOff,
                             (unsigned)(uintptr_t)Bbuf1 + stageOff };

  // async issue of one 16B L piece + one 16B R piece into staging buffer `buf`
  auto issue = [&](int kc, int buf) {
    const void* ga = (const void*)(gA + kc * 32);
    const void* gb = (const void*)(gB + kc * 32);
    asm volatile("global_load_async_to_lds_b128 %0, %1, off"
                 :: "v"(ldsA[buf]), "v"(ga) : "memory");
    asm volatile("global_load_async_to_lds_b128 %0, %1, off"
                 :: "v"(ldsB[buf]), "v"(gb) : "memory");
  };

  issue(0, 0);                                   // prologue: chunk 0 -> buffer 0

  for (int kc = 0; kc < 16; ++kc) {
    const int cur = kc & 1;
    // my wave's async writes for chunk kc have landed in LDS
    asm volatile("s_wait_asynccnt 0x0" ::: "memory");
    // all waves' writes landed; all waves done reading the other buffer
    __syncthreads();
    if (kc + 1 < 16) issue(kc + 1, cur ^ 1);     // overlaps with compute below

    _Float16* As = cur ? Abuf1 : Abuf0;
    _Float16* Bs = cur ? Bbuf1 : Bbuf0;

    v16h bfrag[2];
#pragma unroll
    for (int j = 0; j < 2; ++j) {
      const _Float16* bp = &Bs[((ng * 2 + j) * 16 + lsub) * 40 + hi * 8];
      ((v8h*)&bfrag[j])[0] = *(const v8h*)(bp);
      ((v8h*)&bfrag[j])[1] = *(const v8h*)(bp + 16);
    }
#pragma unroll
    for (int im = 0; im < 4; ++im) {
      const _Float16* ap = &As[((mg * 4 + im) * 16 + lsub) * 40 + hi * 8];
      v16h afrag;
      ((v8h*)&afrag)[0] = *(const v8h*)(ap);
      ((v8h*)&afrag)[1] = *(const v8h*)(ap + 16);
#pragma unroll
      for (int j = 0; j < 2; ++j)
        acc[im][j] = __builtin_amdgcn_wmma_f32_16x16x32_f16(
            false, afrag, false, bfrag[j], (short)0, acc[im][j], false, false);
    }
  }
  __syncthreads();   // all staging reads done before aliased inter overwrite

  // spill inter tile to LDS as f16 [m=256][n=256], row stride 264 halfs
#pragma unroll
  for (int im = 0; im < 4; ++im) {
    int mbase = (mg * 4 + im) * 16;
#pragma unroll
    for (int j = 0; j < 2; ++j) {
      int nn = (ng * 2 + j) * 16 + lsub;
#pragma unroll
      for (int r = 0; r < 8; ++r) {
        int mm = mbase + r + hi * 8;
        inter[mm * 264 + nn] = (_Float16)acc[im][j][r];
      }
    }
  }
  __syncthreads();

  // GEMM2: one 16x16 output tile per wave. rows = pairs p = b_l*8 + d_l
  const int mt2 = wave >> 3, nt2 = wave & 7;
  const int p   = mt2 * 16 + lsub;
  const int b_l = p >> 3, d_l = p & 7;
  v8f acc2 = (v8f){};

  for (int kc = 0; kc < 32; ++kc) {     // K = 1024 = 32 chunks; c = kc, e in [0,32)
    const _Float16* ip = &inter[(b_l * 32 + kc) * 264 + d_l * 32 + hi * 8];
    v16h afrag;
    ((v8h*)&afrag)[0] = *(const v8h*)(ip);
    ((v8h*)&afrag)[1] = *(const v8h*)(ip + 16);
    const _Float16* bp = Wo + (nt2 * 16 + lsub) * 1024 + kc * 32 + hi * 8;
    v16h bfrag;
    ((v8h*)&bfrag)[0] = *(const v8h*)(bp);
    ((v8h*)&bfrag)[1] = *(const v8h*)(bp + 16);
    acc2 = __builtin_amdgcn_wmma_f32_16x16x32_f16(
        false, afrag, false, bfrag, (short)0, acc2, false, false);
  }

  // epilogue: + output_b, / (eps + norm[b,d])
  const int f = nt2 * 16 + lsub;
  const float ob = out_b[f];
#pragma unroll
  for (int r = 0; r < 8; ++r) {
    int pp  = mt2 * 16 + r + hi * 8;
    int b_g = bb * 8 + (pp >> 3);
    int d_g = dd * 8 + (pp & 7);
    float nv = nrm[b_g * NRES + d_g];
    out[((size_t)b_g * NRES + d_g) * CZ + f] = (acc2[r] + ob) / (1e-3f + nv);
  }
}

// ---------------------------------------------------------------------------
extern "C" void kernel_launch(void* const* d_in, const int* in_sizes, int n_in,
                              void* d_out, int out_size, void* d_ws, size_t ws_size,
                              hipStream_t stream) {
  (void)in_sizes; (void)n_in; (void)out_size; (void)ws_size;
  const float* act       = (const float*)d_in[0];
  const float* mask      = (const float*)d_in[1];
  const float* ln_scale  = (const float*)d_in[2];
  const float* ln_offset = (const float*)d_in[3];
  const float* left_w    = (const float*)d_in[4];
  const float* left_b    = (const float*)d_in[5];
  const float* right_w   = (const float*)d_in[6];
  const float* right_b   = (const float*)d_in[7];
  const float* output_w  = (const float*)d_in[8];
  const float* output_b  = (const float*)d_in[9];
  float* out = (float*)d_out;

  char* ws = (char*)d_ws;
  _Float16* Lbuf = (_Float16*)(ws + OFF_L);
  _Float16* Rbuf = (_Float16*)(ws + OFF_R);
  _Float16* Wlr  = (_Float16*)(ws + OFF_WLR);
  _Float16* Wo   = (_Float16*)(ws + OFF_WO);
  float*    nrm  = (float*)(ws + OFF_NORM);

  // 1) weight transposes (f16)
  k_prep<<<(64 * 256 + 128 * 1024 + 255) / 256, 256, 0, stream>>>(
      left_w, right_w, output_w, Wlr, Wo);
  // 2) mask Gram matrix
  k_norm<<<(NRES * NRES + 255) / 256, 256, 0, stream>>>(mask, nrm);
  // 3) LayerNorm + projection (WMMA)
  k_ln_proj<<<dim3(NSEQ / 64, NRES), 128, 0, stream>>>(
      act, mask, ln_scale, ln_offset, left_b, right_b, Wlr, Lbuf, Rbuf);
  // 4) fused outer-product GEMM + output projection (WMMA + async LDS staging)
  constexpr int SMEM_B = 256 * 264 * 2;   // 132 KB dynamic LDS (CDNA5: 320KB/WGP)
  hipFuncSetAttribute((const void*)k_opm,
                      hipFuncAttributeMaxDynamicSharedMemorySize, SMEM_B);
  k_opm<<<dim3(NRES / 8, NRES / 8), 1024, SMEM_B, stream>>>(
      Lbuf, Rbuf, Wo, nrm, output_b, out);
}
